// BF16Indexer_57612691308720
// MI455X (gfx1250) — matrix operand order repaired and checked by reading, not compiled, not run on hardware
//
#include <hip/hip_runtime.h>
#include <hip/hip_bf16.h>

// out[m,n] = sum_h relu( q[m,h,:] . k[n,:] ) * w[m,h] * (1/sqrt(128))
// B=1, M=2048, H=32, D=128, N=4096.  All matmul work on v_wmma_f32_16x16x32_bf16.
// relu(x)*w computed as (x + |x|) * (w*scale/2): |x| is a free src modifier,
// so the epilogue is one v_add_f32 + one v_fmac_f32 per element (no canonicalize).

typedef __attribute__((ext_vector_type(8)))  __bf16 v8bf;
typedef __attribute__((ext_vector_type(16))) __bf16 v16bf;
typedef __attribute__((ext_vector_type(8)))  float  v8f;

#define M_DIM 2048
#define N_DIM 4096
#define H_DIM 32
#define D_DIM 128
#define NT    4     // 16-wide n-tiles per wave  -> wave covers 16x64 output
#define WAVES 4     // waves per block           -> block covers 16x256 output

union AFrag {
    v16bf v;
    struct { v8bf lo; v8bf hi; } p;
};

__global__ __launch_bounds__(WAVES * 32)
void relu_headsum_qk_kernel(const __bf16* __restrict__ q,
                            const __bf16* __restrict__ k,
                            const __bf16* __restrict__ w,
                            float* __restrict__ out)
{
    __shared__ float w_lds[16 * H_DIM];   // weights pre-scaled by scale/2

    const int tid  = threadIdx.x;
    const int wave = tid >> 5;
    const int lane = tid & 31;
    const int l16  = lane & 15;
    const int hi   = lane >> 4;           // half-wave select (ISA fragment layouts)

    const int m0     = blockIdx.y * 16;
    const int n_base = blockIdx.x * (WAVES * NT * 16) + wave * (NT * 16);

    // scale/2 = 1/(2*sqrt(128)); the /2 compensates relu(x) = (x+|x|)/2
    const float half_scale = 0.044194173824159216f;

    // ---- stage scaled weights to LDS: w_lds[mlocal*32 + h] = w[m0+mlocal,h]*scale/2
    {
        const __bf16* wp = w + (size_t)m0 * H_DIM;   // (M,1,H) flat
        for (int i = tid; i < 16 * H_DIM; i += WAVES * 32)
            w_lds[i] = (float)wp[i] * half_scale;
    }
    __syncthreads();

    // ---- preload B fragments (K^T) for this wave's 4 n-tiles, 4 k-steps.
    // B frag (32x16, bf16): lane holds col n = l16, 16 contiguous K values at
    // d = ks*32 + hi*16 .. +15  (32 bytes contiguous in k's row-major layout).
    // Held in registers across all 32 heads (32x traffic reduction on k).
    v16bf bfrag[NT][4];
#pragma unroll
    for (int nt = 0; nt < NT; ++nt) {
        const int n = n_base + nt * 16 + l16;
#pragma unroll
        for (int ks = 0; ks < 4; ++ks) {
            const __bf16* kp = k + (size_t)n * D_DIM + ks * 32 + hi * 16;
            bfrag[nt][ks] = *(const v16bf*)kp;
        }
    }

    v8f acc[NT] = {};   // weighted head-sum accumulators (f32)

    const int mrow = m0 + l16;
    const __bf16* qrow = q + ((size_t)mrow * H_DIM) * D_DIM + hi * 16;

    for (int h = 0; h < H_DIM; ++h) {
        const __bf16* qp = qrow + (size_t)h * D_DIM;

        v8f c[NT] = {};  // per-head logits for the 4 n-tiles

        // A frag (16x32, bf16): lane holds row m = l16,
        // K = ks*32 + hi*16 .. +7 (VGPRs 0-3) and +8..+15 (VGPRs 4-7).
#pragma unroll
        for (int ks = 0; ks < 4; ++ks) {
            AFrag a;
            a.p.lo = *(const v8bf*)(qp + ks * 32);
            a.p.hi = *(const v8bf*)(qp + ks * 32 + 8);
#pragma unroll
            for (int nt = 0; nt < NT; ++nt) {
                c[nt] = __builtin_amdgcn_wmma_f32_16x16x32_bf16(
                    /*neg_a=*/false, a.v,
                    /*neg_b=*/false, bfrag[nt][ks],
                    /*c_mod=*/(short)0, c[nt],
                    /*reuse_a=*/false, /*reuse_b=*/false);
            }
        }

        // Per-row weights: C/D VGPR r, lane group hi -> output row r + 8*hi.
        float wr[8];
#pragma unroll
        for (int r = 0; r < 8; ++r)
            wr[r] = w_lds[(r + 8 * hi) * H_DIM + h];

        // acc += (x + |x|) * (w*scale/2)  ==  relu(x)*w*scale
#pragma unroll
        for (int nt = 0; nt < NT; ++nt)
#pragma unroll
            for (int r = 0; r < 8; ++r) {
                const float x = c[nt][r];
                const float two_relu = x + __builtin_fabsf(x); // v_add_f32 v,|v|,v
                acc[nt][r] = __builtin_fmaf(two_relu, wr[r], acc[nt][r]);
            }
    }

    // ---- store: lane writes rows r+8*hi, col n_base + nt*16 + l16 (coalesced b32)
#pragma unroll
    for (int nt = 0; nt < NT; ++nt) {
        const int n = n_base + nt * 16 + l16;
#pragma unroll
        for (int r = 0; r < 8; ++r)
            out[(size_t)(m0 + r + 8 * hi) * N_DIM + n] = acc[nt][r];
    }
}

extern "C" void kernel_launch(void* const* d_in, const int* in_sizes, int n_in,
                              void* d_out, int out_size, void* d_ws, size_t ws_size,
                              hipStream_t stream) {
    const __bf16* q = (const __bf16*)d_in[0];
    const __bf16* k = (const __bf16*)d_in[1];
    const __bf16* w = (const __bf16*)d_in[2];
    float* out = (float*)d_out;

    dim3 grid(N_DIM / (WAVES * NT * 16), M_DIM / 16);  // (16, 128)
    dim3 block(WAVES * 32);                            // 128 threads = 4 wave32
    relu_headsum_qk_kernel<<<grid, block, 0, stream>>>(q, k, w, out);
}
